// RNN_24378234372233
// MI455X (gfx1250) — compile-verified, matrix operand order checked by reading, not run on hardware
//
#include <hip/hip_runtime.h>

#define VOCAB 50257
#define EMB   256
#define HID   512
#define BATCH 32
#define SEQ   512

typedef __attribute__((ext_vector_type(16))) _Float16     v16h;
typedef __attribute__((ext_vector_type(8)))  float        v8f;
typedef __attribute__((ext_vector_type(4)))  unsigned int u32x4;

// D = A(16x32 f16) * B(32x16 f16) + C(16x16 f32)
static __device__ __forceinline__ v8f wmma_f16(v16h a, v16h b, v8f c) {
  return __builtin_amdgcn_wmma_f32_16x16x32_f16(
      /*neg_a=*/false, a, /*neg_b=*/false, b,
      /*c_mod=*/(short)0, c, /*reuse_a=*/false, /*reuse_b=*/false);
}

// Load one 16x32 f16 WMMA operand fragment from an LDS-resident matrix stored
// row-major ("own-row per lane" form: row = M for A, row = N for B).
// ISA layout: halves j=0..7  -> K = kbase + 8*(lane>>4) + j        (16B chunk)
//             halves j=8..15 -> K = kbase + 8*(lane>>4) + 16 + j-8 (16B chunk)
static __device__ __forceinline__ v16h frag_lds(const _Float16* base, int stride_h,
                                                int row, int kbase, int lane) {
  const int loff = (lane >> 4) << 3;
  const _Float16* p = base + row * stride_h + kbase + loff;
  union { u32x4 u; _Float16 h[8]; } c0, c1;
  c0.u = *(const u32x4*)(p);
  c1.u = *(const u32x4*)(p + 16);
  v16h f;
#pragma unroll
  for (int j = 0; j < 8; ++j) { f[j] = c0.h[j]; f[8 + j] = c1.h[j]; }
  return f;
}

// ---------------------------------------------------------------------------
// Kernel 1: Xpre[t][b][:] = embed[ids[b][t]] @ Wx + bias    (fully parallel)
// One wave per (t, m-tile) row-block; A (gathered embeddings, f16) is built
// once and reused across all 32 N-tiles; Wx is staged transposed in LDS.
// ---------------------------------------------------------------------------
__global__ __launch_bounds__(256) void rnn_xpre_kernel(
    const int* __restrict__ ids, const float* __restrict__ table,
    const float* __restrict__ Wx, const float* __restrict__ bias,
    float* __restrict__ Xpre) {
  extern __shared__ char smem[];
  _Float16* WxT = (_Float16*)smem;                 // [HID][264] (n-major, k<256)
  const int tid = threadIdx.x;

  for (int i = tid; i < EMB * HID; i += 256) {     // stage Wx^T as f16
    const int n = i & (HID - 1);
    const int k = i >> 9;
    WxT[n * 264 + k] = (_Float16)Wx[k * HID + n];
  }
  __syncthreads();

  const int wave = tid >> 5;
  const int lane = tid & 31;
  const int task = blockIdx.x * 8 + wave;          // [0, 2*SEQ)
  const int t    = task >> 1;
  const int mt   = task & 1;
  const int loff = (lane >> 4) << 3;

  const int rowb = (lane & 15) + 16 * mt;          // batch row for this lane
  const int id   = ids[rowb * SEQ + t];

  v16h a[EMB / 32];                                // all A frags for this row-block
#pragma unroll
  for (int kk = 0; kk < EMB / 32; ++kk) {
    v16h f;
#pragma unroll
    for (int j = 0; j < 16; ++j) {
      const int e = kk * 32 + loff + (j & 7) + ((j >> 3) << 4);
      f[j] = (_Float16)table[(long)id * EMB + e];
    }
    a[kk] = f;
  }

  for (int nt = 0; nt < HID / 16; ++nt) {
    const int n = nt * 16 + (lane & 15);
    const float bv = bias[n];
    v8f acc = {bv, bv, bv, bv, bv, bv, bv, bv};
#pragma unroll
    for (int kk = 0; kk < EMB / 32; ++kk)
      acc = wmma_f16(a[kk], frag_lds(WxT, 264, n, kk * 32, lane), acc);
#pragma unroll
    for (int r = 0; r < 8; ++r) {
      const int b = 16 * mt + r + 8 * (lane >> 4); // C layout: M = r + 8*(lane>>4)
      Xpre[(size_t)t * (BATCH * HID) + b * HID + n] = acc[r];
    }
  }
}

// ---------------------------------------------------------------------------
// Kernel 2: the sequential scan. One persistent workgroup (16 waves, 1 WGP).
// Wh(f16) split: K rows 0..255 transposed in LDS (270 KB, conflict-free
// ds_load_b128 frags), K rows 256..511 pinned in VGPRs (128 VGPRs/wave).
// h lives in f32 accumulators per wave (each wave owns 32 columns for all 32
// batch rows) and is republished to LDS as f16 each step.
// Per-step latency trimming:
//   - global_prefetch_b8 of Xpre[t+1] slab + mask col overlaps the WMMA chain
//   - all_h global stores issue AFTER the h-visibility barrier, draining
//     during the next step's compute instead of stalling the barrier.
// ---------------------------------------------------------------------------
__global__ __launch_bounds__(512) void rnn_scan_kernel(
    const float* __restrict__ Wh, const float* __restrict__ Xpre,
    const int* __restrict__ amask, float* __restrict__ all_h,
    float* __restrict__ final_h) {
  extern __shared__ char smem[];
  _Float16* WhT  = (_Float16*)smem;                            // [HID][264], K 0..255
  _Float16* hbuf = (_Float16*)(smem + (size_t)HID * 264 * 2);  // [BATCH][520]

  const int tid  = threadIdx.x;
  const int wave = tid >> 5;
  const int lane = tid & 31;
  const int loff = (lane >> 4) << 3;
  const int nt0  = wave * 2;                       // this wave's two N-tiles

  // Stage Wh^T (K rows 0..255) into LDS as f16.
  for (int i = tid; i < 256 * HID; i += 512) {
    const int n = i & (HID - 1);
    const int k = i >> 9;
    WhT[n * 264 + k] = (_Float16)Wh[k * HID + n];
  }
  // h_{-1} = 0
  for (int i = tid; i < BATCH * 520; i += 512) hbuf[i] = (_Float16)0.f;

  // Register-resident Wh B-fragments for K rows 256..511.
  v16h Breg[2][8];
#pragma unroll
  for (int q = 0; q < 2; ++q) {
    const int n = (nt0 + q) * 16 + (lane & 15);
#pragma unroll
    for (int kk = 0; kk < 8; ++kk) {
      v16h f;
#pragma unroll
      for (int j = 0; j < 16; ++j) {
        const int k = 256 + kk * 32 + loff + (j & 7) + ((j >> 3) << 4);
        f[j] = (_Float16)Wh[k * HID + n];
      }
      Breg[q][kk] = f;
    }
  }

  const v8f vzero = {0.f, 0.f, 0.f, 0.f, 0.f, 0.f, 0.f, 0.f};
  v8f hprev[2][2];                                 // [m-tile][n-tile] f32 state
#pragma unroll
  for (int m = 0; m < 2; ++m)
#pragma unroll
    for (int q = 0; q < 2; ++q) hprev[m][q] = vzero;

  // Warm the first Xpre slab while we wait on staging.
  __builtin_prefetch((const char*)Xpre + tid * 128, 0, 3);
  if (tid < BATCH) __builtin_prefetch(&amask[tid * SEQ], 0, 3);

  __syncthreads();

#pragma unroll 1
  for (int t = 0; t < SEQ; ++t) {
    // acc = Xpre[t] (already contains bias)
    v8f acc[2][2];
#pragma unroll
    for (int m = 0; m < 2; ++m)
#pragma unroll
      for (int q = 0; q < 2; ++q) {
        const int n = (nt0 + q) * 16 + (lane & 15);
        v8f c;
#pragma unroll
        for (int r = 0; r < 8; ++r) {
          const int b = 16 * m + r + 8 * (lane >> 4);
          c[r] = Xpre[(size_t)t * (BATCH * HID) + b * HID + n];
        }
        acc[m][q] = c;
      }

    // Prefetch next step's Xpre slab (512 thr x 128B = 64KB) and mask column;
    // overlaps the entire WMMA + tanh phase below. Speculative, counter-free.
    {
      const int tn = (t + 1 < SEQ) ? (t + 1) : t;
      const char* nxt = (const char*)(Xpre + (size_t)tn * (BATCH * HID));
      __builtin_prefetch(nxt + tid * 128, 0, 3);
      if (tid < BATCH) __builtin_prefetch(&amask[tid * SEQ + tn], 0, 3);
    }

    // K = 0..255 : B from LDS
#pragma unroll
    for (int kk = 0; kk < 8; ++kk) {
      const v16h a0 = frag_lds(hbuf, 520, (lane & 15),      kk * 32, lane);
      const v16h a1 = frag_lds(hbuf, 520, 16 + (lane & 15), kk * 32, lane);
      const v16h b0 = frag_lds(WhT, 264, nt0 * 16 + (lane & 15),       kk * 32, lane);
      const v16h b1 = frag_lds(WhT, 264, (nt0 + 1) * 16 + (lane & 15), kk * 32, lane);
      acc[0][0] = wmma_f16(a0, b0, acc[0][0]);
      acc[1][0] = wmma_f16(a1, b0, acc[1][0]);
      acc[0][1] = wmma_f16(a0, b1, acc[0][1]);
      acc[1][1] = wmma_f16(a1, b1, acc[1][1]);
    }
    // K = 256..511 : B from registers
#pragma unroll
    for (int kk = 0; kk < 8; ++kk) {
      const v16h a0 = frag_lds(hbuf, 520, (lane & 15),      256 + kk * 32, lane);
      const v16h a1 = frag_lds(hbuf, 520, 16 + (lane & 15), 256 + kk * 32, lane);
      acc[0][0] = wmma_f16(a0, Breg[0][kk], acc[0][0]);
      acc[1][0] = wmma_f16(a1, Breg[0][kk], acc[1][0]);
      acc[0][1] = wmma_f16(a0, Breg[1][kk], acc[0][1]);
      acc[1][1] = wmma_f16(a1, Breg[1][kk], acc[1][1]);
    }

    // tanh + mask carry (state stays in registers)
#pragma unroll
    for (int m = 0; m < 2; ++m)
#pragma unroll
      for (int q = 0; q < 2; ++q)
#pragma unroll
        for (int r = 0; r < 8; ++r) {
          const int b = 16 * m + r + 8 * (lane >> 4);
          const float hn = tanhf(acc[m][q][r]);
          const int mv = amask[b * SEQ + t];
          hprev[m][q][r] = mv ? hn : hprev[m][q][r];
        }

    __syncthreads();  // all frag reads of h_{t-1} complete before overwrite

    // Republish h_t as f16 to LDS (the only data the next step depends on).
#pragma unroll
    for (int m = 0; m < 2; ++m)
#pragma unroll
      for (int q = 0; q < 2; ++q) {
        const int n = (nt0 + q) * 16 + (lane & 15);
#pragma unroll
        for (int r = 0; r < 8; ++r) {
          const int b = 16 * m + r + 8 * (lane >> 4);
          hbuf[b * 520 + n] = (_Float16)hprev[m][q][r];
        }
      }

    __syncthreads();  // h_t visible to all waves

    // all_h stores issue after the barrier: they drain while the next step's
    // WMMA chain runs and nothing in this block ever reads them back.
#pragma unroll
    for (int m = 0; m < 2; ++m)
#pragma unroll
      for (int q = 0; q < 2; ++q) {
        const int n = (nt0 + q) * 16 + (lane & 15);
#pragma unroll
        for (int r = 0; r < 8; ++r) {
          const int b = 16 * m + r + 8 * (lane >> 4);
          all_h[(size_t)b * (SEQ * HID) + t * HID + n] = hprev[m][q][r];
        }
      }
  }

  // final hidden state
#pragma unroll
  for (int m = 0; m < 2; ++m)
#pragma unroll
    for (int q = 0; q < 2; ++q) {
      const int n = (nt0 + q) * 16 + (lane & 15);
#pragma unroll
      for (int r = 0; r < 8; ++r) {
        const int b = 16 * m + r + 8 * (lane >> 4);
        final_h[b * HID + n] = hprev[m][q][r];
      }
    }
}

// ---------------------------------------------------------------------------
extern "C" void kernel_launch(void* const* d_in, const int* in_sizes, int n_in,
                              void* d_out, int out_size, void* d_ws, size_t ws_size,
                              hipStream_t stream) {
  const int*   ids   = (const int*)d_in[0];
  const int*   amask = (const int*)d_in[1];
  const float* table = (const float*)d_in[2];
  const float* Wx    = (const float*)d_in[3];
  const float* Wh    = (const float*)d_in[4];
  const float* bias  = (const float*)d_in[5];

  float* all_h   = (float*)d_out;
  float* final_h = all_h + (size_t)BATCH * SEQ * HID;
  float* Xpre    = (float*)d_ws;                 // [SEQ][BATCH][HID] f32 (~32 MB)

  // Parallel pre-GEMM: 1024 wave-tasks (512 t x 2 m-tiles), 8 waves/block.
  const size_t smem_xpre = (size_t)HID * 264 * sizeof(_Float16);      // 270336 B
  rnn_xpre_kernel<<<(2 * SEQ) / 8, 256, smem_xpre, stream>>>(
      ids, table, Wx, bias, Xpre);

  // Latency-critical scan: single persistent workgroup on one WGP.
  const size_t smem_scan = (size_t)HID * 264 * sizeof(_Float16)
                         + (size_t)BATCH * 520 * sizeof(_Float16);    // 303616 B
  rnn_scan_kernel<<<1, 512, smem_scan, stream>>>(
      Wh, Xpre, amask, all_h, final_h);
}
